// CachedGlmExperts_24756191494626
// MI455X (gfx1250) — compile-verified, matrix operand order checked by reading, not run on hardware
//
#include <hip/hip_runtime.h>

// ---------------- problem constants ----------------
#define TT    4096
#define HH    1024
#define II    1408
#define EE    8
#define TOPK  2

// ---------------- tiling ----------------
#define BM    128
#define BN    128
#define BK    32            // K per v_wmma_f32_16x16x32_bf16 step
#define MAXTILES 72         // ceil((T*TOPK + E*(BM-1))/BM) = 72
#define NSLOT (MAXTILES*BM) // 9216 padded slots

typedef __attribute__((ext_vector_type(16))) __bf16 v16bf;
typedef __attribute__((ext_vector_type(8)))  __bf16 v8bf;
typedef __attribute__((ext_vector_type(8)))  float  v8f;
typedef int v4i_vec __attribute__((vector_size(4 * sizeof(int))));

// ---- CDNA5 async global->LDS path (guarded; fallback = manual staging) ----
#if defined(__has_builtin)
#if __has_builtin(__builtin_amdgcn_global_load_async_to_lds_b128)
#define USE_ASYNC 1
#endif
#endif
#ifndef USE_ASYNC
#define USE_ASYNC 0
#endif

#if USE_ASYNC
#if __has_builtin(__builtin_amdgcn_s_wait_asynccnt)
#define WAIT_ASYNC(n) __builtin_amdgcn_s_wait_asynccnt(n)
#else
#define WAIT_ASYNC(n) asm volatile("s_wait_asynccnt %0" ::"i"(n) : "memory")
#endif
__device__ __forceinline__ void async_copy16(const __bf16* g, __bf16* l) {
    // generic->AS(1): full 64-bit address; generic->AS(3): low 32 bits are LDS offset
    __builtin_amdgcn_global_load_async_to_lds_b128(
        (__attribute__((address_space(1))) v4i_vec*)(unsigned long long)g,
        (__attribute__((address_space(3))) v4i_vec*)(unsigned int)(unsigned long long)l,
        0, 0);
}
#else
#define WAIT_ASYNC(n)
#endif

__device__ __forceinline__ __bf16 f2bf(float f) {
    union { float f; unsigned u; } v; v.f = f;
    unsigned r = v.u + 0x7FFFu + ((v.u >> 16) & 1u);   // round-to-nearest-even
    unsigned short s = (unsigned short)(r >> 16);
    __bf16 b; __builtin_memcpy(&b, &s, 2); return b;
}

// ---------------- phase 0: init ----------------
__global__ void init_kernel(int* __restrict__ tokOf, int* __restrict__ counts) {
    int i = blockIdx.x * blockDim.x + threadIdx.x;
    if (i < NSLOT) tokOf[i] = 0;
    if (i < EE)    counts[i] = 0;
}

// ---------------- phase 1: router (softmax + top2 + renorm) ----------------
__global__ void router_kernel(const float* __restrict__ logits,
                              int* __restrict__ counts,
                              int* __restrict__ topkI,
                              float* __restrict__ gates) {
    int t = blockIdx.x * blockDim.x + threadIdx.x;
    if (t >= TT) return;
    float l[EE];
    float mx = -1e30f;
    #pragma unroll
    for (int e = 0; e < EE; ++e) { l[e] = logits[t * EE + e]; mx = fmaxf(mx, l[e]); }
    #pragma unroll
    for (int e = 0; e < EE; ++e) l[e] = __expf(l[e] - mx);
    int i0 = 0;
    #pragma unroll
    for (int e = 1; e < EE; ++e) if (l[e] > l[i0]) i0 = e;
    int i1 = (i0 == 0) ? 1 : 0;
    #pragma unroll
    for (int e = 0; e < EE; ++e) if (e != i0 && l[e] > l[i1]) i1 = e;
    float p0 = l[i0], p1 = l[i1];
    float inv = __builtin_amdgcn_rcpf(p0 + p1);   // softmax denom cancels in renorm
    topkI[t * 2 + 0] = i0;  topkI[t * 2 + 1] = i1;
    gates[t * 2 + 0] = p0 * inv;  gates[t * 2 + 1] = p1 * inv;
    atomicAdd(&counts[i0], 1);
    atomicAdd(&counts[i1], 1);
}

// ---------------- phase 2: plan (padded offsets + fixed tile map) ----------------
__global__ void plan_kernel(const int* __restrict__ counts,
                            int* __restrict__ offsets,
                            int* __restrict__ cursors,
                            int* __restrict__ tileE,
                            int* __restrict__ tileB) {
    if (threadIdx.x != 0 || blockIdx.x != 0) return;
    int off = 0, tc = 0;
    for (int e = 0; e < EE; ++e) {
        offsets[e] = off;
        cursors[e] = 0;
        int tiles = (counts[e] + BM - 1) / BM;
        for (int j = 0; j < tiles && tc < MAXTILES; ++j) {
            tileE[tc] = e; tileB[tc] = off + j * BM; ++tc;
        }
        off += tiles * BM;
    }
    for (; tc < MAXTILES; ++tc) { tileE[tc] = -1; tileB[tc] = 0; }
}

// ---------------- phase 3: scatter tokens to expert slots ----------------
__global__ void scatter_kernel(const int* __restrict__ topkI,
                               const int* __restrict__ offsets,
                               int* __restrict__ cursors,
                               int* __restrict__ tokOf,
                               int* __restrict__ slotOf) {
    int t = blockIdx.x * blockDim.x + threadIdx.x;
    if (t >= TT) return;
    #pragma unroll
    for (int k = 0; k < TOPK; ++k) {
        int e = topkI[t * TOPK + k];
        int s = offsets[e] + atomicAdd(&cursors[e], 1);
        tokOf[s] = t;
        slotOf[t * TOPK + k] = s;
    }
}

// ---------------- phase 4: fp32 -> bf16 conversion ----------------
__global__ void cvt_kernel(const float* __restrict__ src,
                           __bf16* __restrict__ dst, long n) {
    long i = (long)blockIdx.x * blockDim.x + threadIdx.x;
    long stride = (long)gridDim.x * blockDim.x;
    for (; i < n; i += stride) {
        __builtin_prefetch(&src[i + 16384], 0, 1);   // global_prefetch_b8
        dst[i] = f2bf(src[i]);
    }
}

// ---------------- phase 5/6: grouped GEMM with WMMA bf16 ----------------
// MODE 0: A rows gathered via tokOf (x_bf16), SiLU epilogue, bf16 out (h)
// MODE 1: A rows are slots directly (h_bf16), fp32 out (y)
template <int MODE>
__global__ __launch_bounds__(256)
void moe_gemm_kernel(const __bf16* __restrict__ A,
                     const __bf16* __restrict__ Bw,
                     void* __restrict__ Out,
                     const int* __restrict__ tileE,
                     const int* __restrict__ tileB,
                     const int* __restrict__ tokOf,
                     int Kd, int Nd) {
    __shared__ __align__(128) __bf16 ldsA[2][BM][BK];   // 2 x 8 KB
    __shared__ __align__(128) __bf16 ldsB[2][BN][BK];   // 2 x 8 KB

    const int mt = blockIdx.x;
    const int nt = blockIdx.y;
    const int e  = tileE[mt];
    if (e < 0) return;                  // uniform early exit, before any barrier
    const int slot0 = tileB[mt];

    const int tid  = threadIdx.x;
    const int lane = tid & 31;
    const int wid  = tid >> 5;          // 8 waves
    const int wm   = wid & 3;           // 4 waves over M: 32 rows each
    const int wn   = wid >> 2;          // 2 waves over N: 64 cols each
    const int lh   = lane >> 4;         // half-wave select
    const int ln   = lane & 15;

    // cooperative tile loaders: thread -> (row, 16-elem segment)
    const int arow = tid >> 1;          // 0..127
    const int aseg = tid & 1;           // 0..1

    long asrcRow;
    if (MODE == 0) asrcRow = (long)tokOf[slot0 + arow];   // gather token rows
    else           asrcRow = (long)(slot0 + arow);        // compact slot rows
    const __bf16* aptr = A + asrcRow * (long)Kd + aseg * 16;
    const __bf16* bptr = Bw + (long)e * (long)Nd * (long)Kd
                            + ((long)nt * BN + arow) * (long)Kd + aseg * 16;

    auto stage = [&](int kt, int buf) {
        const __bf16* ga = aptr + (long)kt * BK;
        const __bf16* gb = bptr + (long)kt * BK;
        __bf16* la = &ldsA[buf][arow][aseg * 16];
        __bf16* lb = &ldsB[buf][arow][aseg * 16];
#if USE_ASYNC
        async_copy16(ga,     la);
        async_copy16(ga + 8, la + 8);
        async_copy16(gb,     lb);
        async_copy16(gb + 8, lb + 8);
#else
        v8bf a0 = *(const v8bf*)ga;
        v8bf a1 = *(const v8bf*)(ga + 8);
        v8bf b0 = *(const v8bf*)gb;
        v8bf b1 = *(const v8bf*)(gb + 8);
        *(v8bf*)la       = a0;
        *(v8bf*)(la + 8) = a1;
        *(v8bf*)lb       = b0;
        *(v8bf*)(lb + 8) = b1;
#endif
    };

    v8f acc[2][4];
    #pragma unroll
    for (int i = 0; i < 2; ++i)
        #pragma unroll
        for (int j = 0; j < 4; ++j)
            acc[i][j] = (v8f){0.f, 0.f, 0.f, 0.f, 0.f, 0.f, 0.f, 0.f};

    const int ksteps = Kd / BK;
    stage(0, 0);                              // prologue fill of buffer 0
    for (int kt = 0; kt < ksteps; ++kt) {
        const int buf = kt & 1;
        if (kt + 1 < ksteps) {
            stage(kt + 1, buf ^ 1);           // overlap next tile fill w/ compute
            WAIT_ASYNC(4);                    // oldest group (this tile) done
        } else {
            WAIT_ASYNC(0);
        }
        __syncthreads();                      // tile visible to all waves

        v16bf afrag[2], bfrag[4];
        #pragma unroll
        for (int mf = 0; mf < 2; ++mf) {
            int r = wm * 32 + mf * 16 + ln;
            afrag[mf] = *(const v16bf*)&ldsA[buf][r][lh * 16];
        }
        #pragma unroll
        for (int nf = 0; nf < 4; ++nf) {
            int c = wn * 64 + nf * 16 + ln;
            bfrag[nf] = *(const v16bf*)&ldsB[buf][c][lh * 16];
        }
        #pragma unroll
        for (int mf = 0; mf < 2; ++mf)
            #pragma unroll
            for (int nf = 0; nf < 4; ++nf)
                acc[mf][nf] = __builtin_amdgcn_wmma_f32_16x16x32_bf16(
                    false, afrag[mf], false, bfrag[nf],
                    (short)0, acc[mf][nf], false, false);

        __syncthreads();                      // protect buffer reuse (WAR)
    }

    // epilogue: C layout — lane 16h+n holds (M = 8h + r, N = n) in VGPR r
    #pragma unroll
    for (int mf = 0; mf < 2; ++mf) {
        #pragma unroll
        for (int nf = 0; nf < 4; ++nf) {
            const int col     = nt * BN + wn * 64 + nf * 16 + ln;
            const int rowBase = slot0 + wm * 32 + mf * 16 + lh * 8;
            #pragma unroll
            for (int r = 0; r < 8; ++r) {
                float v = acc[mf][nf][r];
                long off = (long)(rowBase + r) * Nd + col;
                if (MODE == 0) {
                    // SiLU via v_exp_f32 + v_rcp_f32 (no IEEE div expansion)
                    float s = v * __builtin_amdgcn_rcpf(1.0f + __expf(-v));
                    ((__bf16*)Out)[off] = f2bf(s);
                } else {
                    ((float*)Out)[off] = v;
                }
            }
        }
    }
}

// ---------------- phase 7: weighted combine (vectorized, no atomics) ----------------
__global__ void combine_kernel(const float* __restrict__ ybuf,
                               const float* __restrict__ gates,
                               const int* __restrict__ slotOf,
                               float* __restrict__ out) {
    long idx = (long)blockIdx.x * blockDim.x + threadIdx.x;   // over T * H/4
    if (idx >= (long)TT * (HH / 4)) return;
    int t  = (int)(idx / (HH / 4));
    int h4 = (int)(idx % (HH / 4));
    int s0 = slotOf[t * 2 + 0], s1 = slotOf[t * 2 + 1];
    float g0 = gates[t * 2 + 0], g1 = gates[t * 2 + 1];
    float4 y0 = ((const float4*)(ybuf + (long)s0 * HH))[h4];
    float4 y1 = ((const float4*)(ybuf + (long)s1 * HH))[h4];
    float4 r;
    r.x = g0 * y0.x + g1 * y1.x;
    r.y = g0 * y0.y + g1 * y1.y;
    r.z = g0 * y0.z + g1 * y1.z;
    r.w = g0 * y0.w + g1 * y1.w;
    ((float4*)out)[idx] = r;
}

// ---------------- launcher ----------------
extern "C" void kernel_launch(void* const* d_in, const int* in_sizes, int n_in,
                              void* d_out, int out_size, void* d_ws, size_t ws_size,
                              hipStream_t stream) {
    (void)in_sizes; (void)n_in; (void)out_size; (void)ws_size;
    const float* x  = (const float*)d_in[0];
    const float* rl = (const float*)d_in[1];
    const float* w1 = (const float*)d_in[2];
    const float* w2 = (const float*)d_in[3];
    float* out = (float*)d_out;
    char* ws = (char*)d_ws;

    size_t o = 0;
    auto alloc = [&](size_t n) {
        o = (o + 255) & ~(size_t)255; size_t r = o; o += n; return r;
    };
    int*    counts  = (int*)(ws + alloc(EE * sizeof(int)));
    int*    cursors = (int*)(ws + alloc(EE * sizeof(int)));
    int*    offsets = (int*)(ws + alloc(EE * sizeof(int)));
    int*    tileE   = (int*)(ws + alloc(MAXTILES * sizeof(int)));
    int*    tileB   = (int*)(ws + alloc(MAXTILES * sizeof(int)));
    int*    topkI   = (int*)(ws + alloc((size_t)TT * TOPK * sizeof(int)));
    float*  gates   = (float*)(ws + alloc((size_t)TT * TOPK * sizeof(float)));
    int*    slotOf  = (int*)(ws + alloc((size_t)TT * TOPK * sizeof(int)));
    int*    tokOf   = (int*)(ws + alloc((size_t)NSLOT * sizeof(int)));
    __bf16* xbf     = (__bf16*)(ws + alloc((size_t)TT * HH * 2));
    __bf16* w1bf    = (__bf16*)(ws + alloc((size_t)EE * II * HH * 2));
    __bf16* w2bf    = (__bf16*)(ws + alloc((size_t)EE * HH * II * 2));
    __bf16* hbuf    = (__bf16*)(ws + alloc((size_t)NSLOT * II * 2));
    float*  ybuf    = (float*)(ws + alloc((size_t)NSLOT * HH * 4));

    init_kernel<<<dim3((NSLOT + 255) / 256), dim3(256), 0, stream>>>(tokOf, counts);
    router_kernel<<<dim3((TT + 255) / 256), dim3(256), 0, stream>>>(rl, counts, topkI, gates);
    plan_kernel<<<dim3(1), dim3(1), 0, stream>>>(counts, offsets, cursors, tileE, tileB);
    scatter_kernel<<<dim3((TT + 255) / 256), dim3(256), 0, stream>>>(topkI, offsets, cursors, tokOf, slotOf);

    cvt_kernel<<<dim3(2048), dim3(256), 0, stream>>>(x,  xbf,  (long)TT * HH);
    cvt_kernel<<<dim3(2048), dim3(256), 0, stream>>>(w1, w1bf, (long)EE * II * HH);
    cvt_kernel<<<dim3(2048), dim3(256), 0, stream>>>(w2, w2bf, (long)EE * HH * II);

    // GEMM1: h = silu(x_gathered @ w1_e^T), N = I, K = H
    moe_gemm_kernel<0><<<dim3(MAXTILES, II / BN), dim3(256), 0, stream>>>(
        xbf, w1bf, (void*)hbuf, tileE, tileB, tokOf, HH, II);
    // GEMM2: y = h @ w2_e^T, N = H, K = I
    moe_gemm_kernel<1><<<dim3(MAXTILES, HH / BN), dim3(256), 0, stream>>>(
        hbuf, w2bf, (void*)ybuf, tileE, tileB, tokOf, II, HH);

    combine_kernel<<<dim3((TT * (HH / 4) + 255) / 256), dim3(256), 0, stream>>>(
        ybuf, gates, slotOf, out);
}
